// SparseEncoder_44770739093787
// MI455X (gfx1250) — compile-verified, matrix operand order checked by reading, not run on hardware
//
#include <hip/hip_runtime.h>
#include <hip/hip_bf16.h>

typedef _Float16 f16;
typedef __attribute__((ext_vector_type(16))) _Float16 v16h;
typedef __attribute__((ext_vector_type(8)))  _Float16 v8h;
typedef __attribute__((ext_vector_type(8)))  float    v8f;

// ---------------------------------------------------------------- utilities
__global__ void zero_f16_kernel(f16* p, int n) {
    int i = blockIdx.x * blockDim.x + threadIdx.x;
    if (i < n) p[i] = (f16)0.f;
}
__global__ void zero_f32_kernel(float* p, int n) {
    int i = blockIdx.x * blockDim.x + threadIdx.x;
    if (i < n) p[i] = 0.f;
}

// Pack f32 weights (3,3,3,CI,CO) into the per-lane WMMA B-fragment layout:
//   wp[((kchunk*CO + n)*2 + half)*16 + j] = W[kk*CO + n],  kk = kchunk*32 + ko(half,j)
// zero-padded where kk >= K, so the conv K-loop needs no ragged tail.
__global__ void pack_weights_kernel(const float* __restrict__ w, f16* __restrict__ wp,
                                    int K, int CO, int total) {
    int i = blockIdx.x * blockDim.x + threadIdx.x;
    if (i >= total) return;
    int j       = i & 15;
    int halfIdx = (i >> 4) & 1;
    int n       = (i >> 5) % CO;
    int kchunk  = (i >> 5) / CO;
    int ko = (j < 8) ? (halfIdx * 8 + j) : (16 + halfIdx * 8 + (j - 8));
    int kk = kchunk * 32 + ko;
    wp[i] = (kk < K) ? (f16)w[kk * CO + n] : (f16)0.f;
}

// scatter voxels (N_VOX x 4) into dense (21,200,176,4) f16 grid + f32 mask
__global__ void scatter_kernel(const float* __restrict__ vox, const int* __restrict__ coors,
                               f16* __restrict__ act, float* __restrict__ mask, int n) {
    int i = blockIdx.x * blockDim.x + threadIdx.x;
    if (i >= n) return;
    int z = coors[i * 4 + 1], y = coors[i * 4 + 2], x = coors[i * 4 + 3];
    int v = (z * 200 + y) * 176 + x;
    mask[v] = 1.0f;
#pragma unroll
    for (int c = 0; c < 4; ++c) act[v * 4 + c] = (f16)vox[i * 4 + c];
}

// mask dilation = reduce_window(max, 3x3x3, stride s, pad 1)
__global__ void dilate_kernel(const float* __restrict__ mi, float* __restrict__ mo,
                              int Di, int Hi, int Wi, int Do, int Ho, int Wo,
                              int sd, int sh, int sw) {
    int m = blockIdx.x * blockDim.x + threadIdx.x;
    int M = Do * Ho * Wo;
    if (m >= M) return;
    int od = m / (Ho * Wo);
    int r  = m - od * (Ho * Wo);
    int oh = r / Wo;
    int ow = r - oh * Wo;
    float v = 0.f;
    for (int kd = 0; kd < 3; ++kd) {
        int id = od * sd + kd - 1;
        if ((unsigned)id >= (unsigned)Di) continue;
        for (int kh = 0; kh < 3; ++kh) {
            int ih = oh * sh + kh - 1;
            if ((unsigned)ih >= (unsigned)Hi) continue;
            for (int kw = 0; kw < 3; ++kw) {
                int iw = ow * sw + kw - 1;
                if ((unsigned)iw >= (unsigned)Wi) continue;
                v = fmaxf(v, mi[(id * Hi + ih) * Wi + iw]);
            }
        }
    }
    mo[m] = v;
}

// ------------------------------------------------ implicit-GEMM conv (WMMA)
// One wave computes a full 16 x CO strip. Per K-chunk: A fragment = two
// predicated aligned v8h loads (channels are contiguous within a tap group),
// B fragment = one aligned v16h load from the pre-packed weight buffer,
// then CO/16 back-to-back v_wmma_f32_16x16x32_f16.
template <int CI, int CO>
__global__ __launch_bounds__(256) void conv_wmma_kernel(
    const f16* __restrict__ in, const f16* __restrict__ wp, f16* __restrict__ out,
    int Di, int Hi, int Wi, int Do, int Ho, int Wo,
    int sd, int sh, int sw, int M, int Mtiles) {

    constexpr int K  = 27 * CI;
    constexpr int LC = (CI == 4) ? 2 : (CI == 16) ? 4 : (CI == 32) ? 5 : 6;
    constexpr int NT = CO / 16;
    constexpr int CHUNKS = (K + 31) / 32;

    __shared__ int taps[256 * 32];                 // 27 taps + 5 pad sentinels

    int wave = blockIdx.x * (blockDim.x >> 5) + (threadIdx.x >> 5);
    if (wave >= Mtiles) return;                    // wave-uniform: EXEC stays all-1s
    int lane = threadIdx.x & 31;
    int row  = lane & 15;
    int half = lane >> 4;
    int* myTaps = &taps[threadIdx.x * 32];

    // A-row -> output voxel coordinates
    int m = wave * 16 + row;
    bool valid = (m < M);
    int od = 0, oh = 0, ow = 0;
    if (valid) {
        od = m / (Ho * Wo);
        int r = m - od * (Ho * Wo);
        oh = r / Wo;
        ow = r - oh * Wo;
    }
    // 27 neighbor base offsets (element index into `in`, -1 = zero pad)
#pragma unroll
    for (int kd = 0; kd < 3; ++kd)
#pragma unroll
        for (int kh = 0; kh < 3; ++kh)
#pragma unroll
            for (int kw = 0; kw < 3; ++kw) {
                int id = od * sd + kd - 1, ih = oh * sh + kh - 1, iw = ow * sw + kw - 1;
                int off = -1;
                if (valid && (unsigned)id < (unsigned)Di &&
                    (unsigned)ih < (unsigned)Hi && (unsigned)iw < (unsigned)Wi) {
                    off = (((id * Hi) + ih) * Wi + iw) * CI;
                    __builtin_prefetch(in + off, 0, 1);   // global_prefetch_b8
                }
                myTaps[(kd * 3 + kh) * 3 + kw] = off;
            }
#pragma unroll
    for (int t = 27; t < 32; ++t) myTaps[t] = -1;  // zero-padded K region

    v8f zero = {};
    v8f acc[NT];
#pragma unroll
    for (int t = 0; t < NT; ++t) acc[t] = zero;

    const f16* bbase = wp + ((size_t)row * 2 + half) * 16;  // chunk 0, n = row
    for (int kc8 = 0; kc8 < CHUNKS; ++kc8) {
        int kc = kc8 * 32;
        v16h a;
        if constexpr (CI >= 16) {
            // each 8-element group lies in one tap with contiguous channels
            int kk0 = kc + half * 8;
            int kk1 = kc + 16 + half * 8;
            int off0 = myTaps[kk0 >> LC];
            int off1 = myTaps[kk1 >> LC];
            v8h zero8 = {};
            v8h lo = (off0 >= 0) ? *(const v8h*)(in + off0 + (kk0 & (CI - 1))) : zero8;
            v8h hi = (off1 >= 0) ? *(const v8h*)(in + off1 + (kk1 & (CI - 1))) : zero8;
#pragma unroll
            for (int j = 0; j < 8; ++j) { a[j] = lo[j]; a[j + 8] = hi[j]; }
        } else {                                   // CI = 4 (layer 0 only)
#pragma unroll
            for (int j = 0; j < 16; ++j) {
                int kk = kc + ((j < 8) ? (half * 8 + j) : (16 + half * 8 + (j - 8)));
                int off = myTaps[kk >> LC];
                a[j] = (off >= 0) ? in[off + (kk & (CI - 1))] : (f16)0.f;
            }
        }
        const f16* bp = bbase + (size_t)kc8 * CO * 32;
#pragma unroll
        for (int t = 0; t < NT; ++t) {
            v16h b = *(const v16h*)(bp + (size_t)t * 512);   // n += 16 -> +16*32 elems
            acc[t] = __builtin_amdgcn_wmma_f32_16x16x32_f16(
                false, a, false, b, (short)0, acc[t], false, false);
        }
    }

    // C/D layout: M = 8*half + vgpr, N = lane&15
#pragma unroll
    for (int t = 0; t < NT; ++t)
#pragma unroll
        for (int v = 0; v < 8; ++v) {
            int mo = wave * 16 + half * 8 + v;
            if (mo < M) out[mo * CO + t * 16 + row] = (f16)acc[t][v];
        }
}

// ------------------------------------- masked BN statistics: S1, S2, count
// stats layout: [0..127]=sum(x*m), [128..255]=sum(x^2*m), [256]=sum(m)
__global__ __launch_bounds__(256) void stats_kernel(
    const f16* __restrict__ x, const float* __restrict__ mask,
    float* __restrict__ stats, int nElem, int lcC) {
    __shared__ float s1[128], s2[128];
    __shared__ float scnt;
    int C = 1 << lcC;
    int tid = threadIdx.x;
    if (tid < C) { s1[tid] = 0.f; s2[tid] = 0.f; }
    if (tid == 0) scnt = 0.f;
    __syncthreads();
    int stride = blockDim.x * gridDim.x;
    for (int i = blockIdx.x * blockDim.x + tid; i < nElem; i += stride) {
        int vox = i >> lcC;
        int c   = i & (C - 1);
        float mv = mask[vox];
        if (mv != 0.f) {
            float v = (float)x[i];
            atomicAdd(&s1[c], v);
            atomicAdd(&s2[c], v * v);
            if (c == 0) atomicAdd(&scnt, mv);
        }
    }
    __syncthreads();
    if (tid < C) {
        atomicAdd(&stats[tid], s1[tid]);
        atomicAdd(&stats[128 + tid], s2[tid]);
    }
    if (tid == 0) atomicAdd(&stats[256], scnt);
}

// -------------------------------- fused BN + exact GELU + mask (in place)
__global__ void bn_gelu_kernel(f16* __restrict__ x, const float* __restrict__ mask,
                               const float* __restrict__ stats,
                               const float* __restrict__ gamma, const float* __restrict__ beta,
                               int nElem, int lcC, float* __restrict__ finalOut) {
    int i = blockIdx.x * blockDim.x + threadIdx.x;
    if (i >= nElem) return;
    int C = 1 << lcC;
    int c   = i & (C - 1);
    int vox = i >> lcC;
    float cnt  = fmaxf(stats[256], 1.0f);
    float mean = stats[c] / cnt;
    float var  = stats[128 + c] / cnt - mean * mean;
    float rstd = rsqrtf(var + 1e-5f);
    float v = (float)x[i];
    float y = (v - mean) * rstd * gamma[c] + beta[c];
    float g = 0.5f * y * (1.0f + erff(y * 0.70710678118654752f));
    float r = g * mask[vox];
    if (finalOut) finalOut[i] = r;
    else          x[i] = (f16)r;
}

__global__ void add_f16_kernel(f16* __restrict__ x, const f16* __restrict__ y, int n) {
    int i = blockIdx.x * blockDim.x + threadIdx.x;
    if (i < n) x[i] = (f16)((float)x[i] + (float)y[i]);
}

// ---------------------------------------------------------------- launcher
extern "C" void kernel_launch(void* const* d_in, const int* in_sizes, int n_in,
                              void* d_out, int out_size, void* d_ws, size_t ws_size,
                              hipStream_t stream) {
    const float* voxels = (const float*)d_in[0];
    const int*   coors  = (const int*)d_in[1];
    const float* W[12]; const float* G[12]; const float* Bt[12];
    for (int i = 0; i < 12; ++i) {
        W[i]  = (const float*)d_in[2 + i];
        G[i]  = (const float*)d_in[14 + i];
        Bt[i] = (const float*)d_in[26 + i];
    }

    static const int CI[12] = {4, 16, 16, 32, 32, 32, 64, 64, 64, 64, 64, 64};
    static const int CO[12] = {16, 16, 32, 32, 32, 64, 64, 64, 64, 64, 64, 128};
    const int V0 = 21 * 200 * 176;  // 739200
    const int V1 = 11 * 100 * 88;   // 96800
    const int V2 = 6 * 50 * 44;     // 13200
    const int V3 = 3 * 25 * 22;     // 1650
    const int V4 = 2 * 25 * 22;     // 1100

    // carve workspace (~60 MB)
    char* p = (char*)d_ws;
    auto alloc = [&](size_t bytes) -> char* {
        char* r = p;
        p += (bytes + 255) & ~(size_t)255;
        return r;
    };
    f16* actA   = (f16*)alloc((size_t)V0 * 16 * sizeof(f16));
    f16* actB   = (f16*)alloc((size_t)V0 * 16 * sizeof(f16));
    f16* idn    = (f16*)alloc((size_t)V1 * 32 * sizeof(f16));
    float* mask0 = (float*)alloc((size_t)V0 * 4);
    float* mask1 = (float*)alloc((size_t)V1 * 4);
    float* mask2 = (float*)alloc((size_t)V2 * 4);
    float* mask3 = (float*)alloc((size_t)V3 * 4);
    float* mask4 = (float*)alloc((size_t)V4 * 4);
    float* stats = (float*)alloc(272 * 4);
    f16* wts[12];
    int  wpsz[12];
    for (int i = 0; i < 12; ++i) {
        int chunks = (27 * CI[i] + 31) / 32;
        wpsz[i] = chunks * CO[i] * 32;             // padded/packed element count
        wts[i] = (f16*)alloc((size_t)wpsz[i] * sizeof(f16));
    }

    auto gridFor = [](int n) { return (unsigned)((n + 255) / 256); };

    // init: zero input grid + mask, scatter voxels, pack weights to fragment layout
    zero_f16_kernel<<<gridFor(V0 * 4), 256, 0, stream>>>(actA, V0 * 4);
    zero_f32_kernel<<<gridFor(V0), 256, 0, stream>>>(mask0, V0);
    scatter_kernel<<<gridFor(80000), 256, 0, stream>>>(voxels, coors, actA, mask0, 80000);
    for (int i = 0; i < 12; ++i)
        pack_weights_kernel<<<gridFor(wpsz[i]), 256, 0, stream>>>(
            W[i], wts[i], 27 * CI[i], CO[i], wpsz[i]);

    auto conv = [&](const f16* in, f16* out, int L,
                    int Di, int Hi, int Wi, int Do, int Ho, int Wo,
                    int sd, int sh, int sw) {
        int M = Do * Ho * Wo;
        int Mt = (M + 15) / 16;
        int blocks = (Mt + 7) / 8;  // 8 waves / block (wave32)
        f16* w = wts[L];
#define CONV_CASE(ci, co)                                                        \
    conv_wmma_kernel<ci, co><<<blocks, 256, 0, stream>>>(                        \
        in, w, out, Di, Hi, Wi, Do, Ho, Wo, sd, sh, sw, M, Mt)
        switch (L) {
            case 0: CONV_CASE(4, 16); break;
            case 1: CONV_CASE(16, 16); break;
            case 2: CONV_CASE(16, 32); break;
            case 3: case 4: CONV_CASE(32, 32); break;
            case 5: CONV_CASE(32, 64); break;
            case 6: case 7: case 8: case 9: case 10: CONV_CASE(64, 64); break;
            case 11: CONV_CASE(64, 128); break;
        }
#undef CONV_CASE
    };
    auto bn = [&](f16* x, const float* mask, int L, int Mvox, float* fout) {
        int co = CO[L];
        int lcC = (co == 16) ? 4 : (co == 32) ? 5 : (co == 64) ? 6 : 7;
        int nE = Mvox * co;
        zero_f32_kernel<<<gridFor(272), 256, 0, stream>>>(stats, 272);
        int sb = (nE + 255) / 256;
        if (sb > 2048) sb = 2048;
        stats_kernel<<<sb, 256, 0, stream>>>(x, mask, stats, nE, lcC);
        bn_gelu_kernel<<<gridFor(nE), 256, 0, stream>>>(x, mask, stats, G[L], Bt[L], nE, lcC, fout);
    };

    // L0, L1: full-res subm, 4->16->16
    conv(actA, actB, 0, 21, 200, 176, 21, 200, 176, 1, 1, 1);
    bn(actB, mask0, 0, V0, nullptr);
    conv(actB, actA, 1, 21, 200, 176, 21, 200, 176, 1, 1, 1);
    bn(actA, mask0, 1, V0, nullptr);

    // block 1: L2 stride2 (16->32) + L3,L4 + residual
    conv(actA, actB, 2, 21, 200, 176, 11, 100, 88, 2, 2, 2);
    dilate_kernel<<<gridFor(V1), 256, 0, stream>>>(mask0, mask1, 21, 200, 176, 11, 100, 88, 2, 2, 2);
    bn(actB, mask1, 2, V1, nullptr);
    hipMemcpyAsync(idn, actB, (size_t)V1 * 32 * sizeof(f16), hipMemcpyDeviceToDevice, stream);
    conv(actB, actA, 3, 11, 100, 88, 11, 100, 88, 1, 1, 1);
    bn(actA, mask1, 3, V1, nullptr);
    conv(actA, actB, 4, 11, 100, 88, 11, 100, 88, 1, 1, 1);
    bn(actB, mask1, 4, V1, nullptr);
    add_f16_kernel<<<gridFor(V1 * 32), 256, 0, stream>>>(actB, idn, V1 * 32);

    // block 2: L5 stride2 (32->64) + L6,L7 + residual
    conv(actB, actA, 5, 11, 100, 88, 6, 50, 44, 2, 2, 2);
    dilate_kernel<<<gridFor(V2), 256, 0, stream>>>(mask1, mask2, 11, 100, 88, 6, 50, 44, 2, 2, 2);
    bn(actA, mask2, 5, V2, nullptr);
    hipMemcpyAsync(idn, actA, (size_t)V2 * 64 * sizeof(f16), hipMemcpyDeviceToDevice, stream);
    conv(actA, actB, 6, 6, 50, 44, 6, 50, 44, 1, 1, 1);
    bn(actB, mask2, 6, V2, nullptr);
    conv(actB, actA, 7, 6, 50, 44, 6, 50, 44, 1, 1, 1);
    bn(actA, mask2, 7, V2, nullptr);
    add_f16_kernel<<<gridFor(V2 * 64), 256, 0, stream>>>(actA, idn, V2 * 64);

    // block 3: L8 stride2 (64->64) + L9,L10 + residual
    conv(actA, actB, 8, 6, 50, 44, 3, 25, 22, 2, 2, 2);
    dilate_kernel<<<gridFor(V3), 256, 0, stream>>>(mask2, mask3, 6, 50, 44, 3, 25, 22, 2, 2, 2);
    bn(actB, mask3, 8, V3, nullptr);
    hipMemcpyAsync(idn, actB, (size_t)V3 * 64 * sizeof(f16), hipMemcpyDeviceToDevice, stream);
    conv(actB, actA, 9, 3, 25, 22, 3, 25, 22, 1, 1, 1);
    bn(actA, mask3, 9, V3, nullptr);
    conv(actA, actB, 10, 3, 25, 22, 3, 25, 22, 1, 1, 1);
    bn(actB, mask3, 10, V3, nullptr);
    add_f16_kernel<<<gridFor(V3 * 64), 256, 0, stream>>>(actB, idn, V3 * 64);

    // L11: stride (2,1,1), 64->128; BN+GELU writes f32 directly into d_out
    conv(actB, actA, 11, 3, 25, 22, 2, 25, 22, 2, 1, 1);
    dilate_kernel<<<gridFor(V4), 256, 0, stream>>>(mask3, mask4, 3, 25, 22, 2, 25, 22, 2, 1, 1);
    bn(actA, mask4, 11, V4, (float*)d_out);
}
